// Transformer_5171140624905
// MI455X (gfx1250) — matrix operand order repaired and checked by reading, not compiled, not run on hardware
//
#include <hip/hip_runtime.h>
#include <math.h>

// ---------------------------------------------------------------------------
// MI455X (gfx1250) transformer block: bf16 WMMA, fully async LDS staging,
// all transposes hoisted out of hot loops.
// ---------------------------------------------------------------------------

typedef __bf16 bf16_t;
typedef __attribute__((ext_vector_type(8)))  __bf16 v8bf;
typedef __attribute__((ext_vector_type(16))) __bf16 v16bf;
typedef __attribute__((ext_vector_type(8)))  float  v8f;
typedef __attribute__((ext_vector_type(4)))  int    v4i;

#define B_    256
#define T_    200
#define E_    512
#define H_    8
#define DH_   64
#define TPAD  224          // T padded to 14 x 16 tiles
#define M_    (B_ * T_)    // 51200 rows, divisible by 128
#define FF_   2048
#define NEGV  (-4294967295.0f)
#define SQRT_E 22.627416997969522f

// ---- CDNA5 async global->LDS (ASYNCcnt) with safe fallback ----------------
#if defined(__has_builtin)
#if __has_builtin(__builtin_amdgcn_global_load_async_to_lds_b128)
#define HAVE_ASYNC_LDS 1
#endif
#endif
#ifndef HAVE_ASYNC_LDS
#define HAVE_ASYNC_LDS 0
#endif

// Copy 8 bf16 (16B) global -> LDS.
__device__ __forceinline__ void stage8(const bf16_t* gsrc, bf16_t* ldst) {
#if HAVE_ASYNC_LDS
    __builtin_amdgcn_global_load_async_to_lds_b128(
        (__attribute__((address_space(1))) v4i*)(v4i*)gsrc,
        (__attribute__((address_space(3))) v4i*)(v4i*)ldst, 0, 0);
#else
    *(uint4*)ldst = *(const uint4*)gsrc;
#endif
}

// Copy 16 bf16 (32B) global -> LDS.
__device__ __forceinline__ void stage16(const bf16_t* gsrc, bf16_t* ldst) {
    stage8(gsrc, ldst);
    stage8(gsrc + 8, ldst + 8);
}

__device__ __forceinline__ void wait_async_lds() {
#if HAVE_ASYNC_LDS
#if __has_builtin(__builtin_amdgcn_s_wait_asynccnt)
    __builtin_amdgcn_s_wait_asynccnt(0);
#else
    asm volatile("s_wait_asynccnt 0" ::: "memory");
#endif
#endif
}

// ---- WMMA helpers ---------------------------------------------------------
__device__ __forceinline__ v16bf cat16(v8bf lo, v8bf hi) {
    return __builtin_shufflevector(lo, hi, 0,1,2,3,4,5,6,7,8,9,10,11,12,13,14,15);
}

__device__ __forceinline__ v8f wmma_bf16(v16bf a, v16bf b, v8f c) {
    return __builtin_amdgcn_wmma_f32_16x16x32_bf16(false, a, false, b, (short)0, c, false, false);
}

// A fragment (16x32 tile at (m0,k0)) from row-major [*, lda] bf16.
__device__ __forceinline__ v16bf load_frag_a(const bf16_t* A, int lda, int m0, int k0, int lane) {
    int m  = m0 + (lane & 15);
    int kb = k0 + ((lane >> 4) << 3);
    v8bf lo = *(const v8bf*)(A + m * lda + kb);
    v8bf hi = *(const v8bf*)(A + m * lda + kb + 16);
    return cat16(lo, hi);
}

// B fragment (32x16, KxN) from K-transposed storage Bt[n][ldb].
__device__ __forceinline__ v16bf load_frag_bt(const bf16_t* Bt, int ldb, int n0, int k0, int lane) {
    int n  = n0 + (lane & 15);
    int kb = k0 + ((lane >> 4) << 3);
    v8bf lo = *(const v8bf*)(Bt + n * ldb + kb);
    v8bf hi = *(const v8bf*)(Bt + n * ldb + kb + 16);
    return cat16(lo, hi);
}

// ---------------------------------------------------------------------------
// Kernel 1: convert weights f32 -> bf16, TRANSPOSED to [N][K]
// ---------------------------------------------------------------------------
__global__ void prep_weights(const float* __restrict__ wq, const float* __restrict__ wk,
                             const float* __restrict__ wv, const float* __restrict__ f1,
                             const float* __restrict__ f2,
                             bf16_t* __restrict__ wqt, bf16_t* __restrict__ wkt,
                             bf16_t* __restrict__ wvt, bf16_t* __restrict__ f1t,
                             bf16_t* __restrict__ f2t) {
    const int EW = E_ * E_;
    const int FW = E_ * FF_;
    const int total = 3 * EW + 2 * FW;
    for (int i = blockIdx.x * blockDim.x + threadIdx.x; i < total; i += gridDim.x * blockDim.x) {
        int j = i;
        if (j < EW) { int k = j / E_,  n = j % E_;  wqt[n * E_  + k] = (bf16_t)wq[j]; continue; }
        j -= EW;
        if (j < EW) { int k = j / E_,  n = j % E_;  wkt[n * E_  + k] = (bf16_t)wk[j]; continue; }
        j -= EW;
        if (j < EW) { int k = j / E_,  n = j % E_;  wvt[n * E_  + k] = (bf16_t)wv[j]; continue; }
        j -= EW;
        if (j < FW) { int k = j / FF_, n = j % FF_; f1t[n * E_  + k] = (bf16_t)f1[j]; continue; }
        j -= FW;
        {             int k = j / E_,  n = j % E_;  f2t[n * FF_ + k] = (bf16_t)f2[j]; }
    }
}

// ---------------------------------------------------------------------------
// Kernel 2: q_in/k_in = x + pos * sqrt(E), stored bf16
// ---------------------------------------------------------------------------
__global__ void prep_inputs(const float* __restrict__ q, const float* __restrict__ k,
                            const float* __restrict__ pq, const float* __restrict__ pk,
                            bf16_t* __restrict__ qb, bf16_t* __restrict__ kb) {
    const int n = M_ * E_;
    for (int i = blockIdx.x * blockDim.x + threadIdx.x; i < n; i += gridDim.x * blockDim.x) {
        int p = i % (T_ * E_);
        qb[i] = (bf16_t)(q[i] + pq[p] * SQRT_E);
        kb[i] = (bf16_t)(k[i] + pk[p] * SQRT_E);
    }
}

// ---------------------------------------------------------------------------
// Shared 128x64 GEMM pipeline (double-buffered LDS, all-async staging).
//   A:  row-major bf16 [*, lda]
//   Bt: pre-transposed bf16 [N][ldk]  (contiguous K per output column)
// ---------------------------------------------------------------------------
#define GEMM_PIPELINE(Aptr, lda, Btptr, ldk, KDIM)                                  \
    const int tid = threadIdx.x, lane = tid & 31, wave = tid >> 5;                  \
    const int wr = wave & 3, wc = wave >> 2;                                        \
    const int arow = tid >> 1, acb = (tid & 1) * 16;                                \
    const int bn = tid >> 2, bkc = (tid & 3) * 8;                                   \
    v8f acc[2][2] = {};                                                             \
    const int NK = (KDIM) / 32;                                                     \
    stage16((Aptr) + (size_t)(rowBase + arow) * (lda) + acb,                        \
            &As[0][arow * 32 + acb]);                                               \
    stage8((Btptr) + (size_t)(nBase + bn) * (ldk) + bkc,                            \
           &Bs[0][bn * 32 + bkc]);                                                  \
    wait_async_lds();                                                               \
    __syncthreads();                                                                \
    for (int i = 0; i < NK; i++) {                                                  \
        const int cur = i & 1;                                                      \
        if (i + 1 < NK) {   /* stage next k-step into the other buffer */           \
            const int k0n = (i + 1) * 32;                                           \
            stage16((Aptr) + (size_t)(rowBase + arow) * (lda) + k0n + acb,          \
                    &As[cur ^ 1][arow * 32 + acb]);                                 \
            stage8((Btptr) + (size_t)(nBase + bn) * (ldk) + k0n + bkc,              \
                   &Bs[cur ^ 1][bn * 32 + bkc]);                                    \
        }                                                                           \
        if (i + 2 < NK)                                                             \
            __builtin_prefetch((Btptr) + (size_t)(nBase + bn) * (ldk)               \
                               + (i + 2) * 32 + bkc, 0, 3);                         \
        _Pragma("unroll")                                                           \
        for (int sm = 0; sm < 2; sm++) {                                            \
            v16bf a = load_frag_a(&As[cur][0], 32, wr * 32 + sm * 16, 0, lane);     \
            _Pragma("unroll")                                                       \
            for (int sn = 0; sn < 2; sn++) {                                        \
                v16bf b = load_frag_bt(&Bs[cur][0], 32, wc * 32 + sn * 16, 0, lane);\
                acc[sm][sn] = wmma_bf16(a, b, acc[sm][sn]);                         \
            }                                                                       \
        }                                                                           \
        wait_async_lds();                                                           \
        __syncthreads();                                                            \
    }

// ---------------------------------------------------------------------------
// Kernel 3: projections  Q/K/V = in @ W   (M=51200, N=512, K=512)
//   Q,K stored [B,H,TPAD,DH]; V stored transposed [B,H,DH,TPAD].
// ---------------------------------------------------------------------------
__global__ __launch_bounds__(256) void proj_gemm(
        const bf16_t* __restrict__ qin, const bf16_t* __restrict__ kin,
        const bf16_t* __restrict__ wqt, const bf16_t* __restrict__ wkt,
        const bf16_t* __restrict__ wvt,
        bf16_t* __restrict__ Qh, bf16_t* __restrict__ Kh, bf16_t* __restrict__ VtG) {
    __shared__ __align__(16) bf16_t As[2][128 * 32];
    __shared__ __align__(16) bf16_t Bs[2][64 * 32];

    const int which = blockIdx.z;
    const bf16_t* A  = (which == 0) ? qin : kin;
    const bf16_t* Wt = (which == 0) ? wqt : (which == 1) ? wkt : wvt;
    bf16_t* O        = (which == 0) ? Qh  : (which == 1) ? Kh  : VtG;
    const float scale = (which == 0) ? 0.125f : 1.0f;   // 1/sqrt(Dh)
    const int rowBase = blockIdx.y * 128;
    const int h = blockIdx.x;
    const int nBase = h * 64;

    GEMM_PIPELINE(A, E_, Wt, E_, E_)

    #pragma unroll
    for (int sm = 0; sm < 2; sm++)
        #pragma unroll
        for (int sn = 0; sn < 2; sn++)
            #pragma unroll
            for (int v = 0; v < 8; v++) {
                int r = rowBase + wr * 32 + sm * 16 + v + ((lane >> 4) << 3);
                int d = wc * 32 + sn * 16 + (lane & 15);
                int bb = r / T_, t = r % T_;
                bf16_t val = (bf16_t)(acc[sm][sn][v] * scale);
                if (which == 2)   // V transposed: [B,H,DH,TPAD]
                    O[(size_t)((bb * H_ + h) * DH_ + d) * TPAD + t] = val;
                else              // Q,K: [B,H,TPAD,DH]
                    O[(size_t)((bb * H_ + h) * TPAD + t) * DH_ + d] = val;
            }
}

// ---------------------------------------------------------------------------
// Kernel 4: attention per (b, h, 32-row query tile).  128 threads = 4 waves.
//   V is pre-transposed in global -> phase-3 B fragments load straight from HBM.
// ---------------------------------------------------------------------------
__global__ __launch_bounds__(128) void attn_kernel(
        const bf16_t* __restrict__ Qh, const bf16_t* __restrict__ Kh,
        const bf16_t* __restrict__ VtG,
        const int* __restrict__ qlen, const int* __restrict__ klen,
        const float* __restrict__ queries, const float* __restrict__ pos_q,
        float* __restrict__ out1, bf16_t* __restrict__ out1b) {
    __shared__ __align__(16) char smem[28672 + 14336];
    float*  scores = (float*)smem;            // [32][224] f32
    bf16_t* attn   = (bf16_t*)(smem + 28672); // [32][224] bf16

    const int rt = blockIdx.x, h = blockIdx.y, b = blockIdx.z;
    const int row0 = rt * 32;
    const int tid = threadIdx.x, lane = tid & 31, wave = tid >> 5;
    const bf16_t* Qp = Qh  + (size_t)((b * H_ + h) * TPAD) * DH_;
    const bf16_t* Kp = Kh  + (size_t)((b * H_ + h) * TPAD) * DH_;
    const bf16_t* Vp = VtG + (size_t)((b * H_ + h) * DH_) * TPAD;

    // ---- phase 1: scores = (Q/sqrt(Dh)) K^T ----
    {
        const int mstrip = wave & 1;
        v16bf qf0 = load_frag_a(Qp, DH_, row0 + mstrip * 16, 0,  lane);
        v16bf qf1 = load_frag_a(Qp, DH_, row0 + mstrip * 16, 32, lane);
        for (int ct = (wave >> 1); ct < 13; ct += 2) {
            v16bf b0 = load_frag_bt(Kp, DH_, ct * 16, 0,  lane);
            v16bf b1 = load_frag_bt(Kp, DH_, ct * 16, 32, lane);
            v8f s = {};
            s = wmma_bf16(qf0, b0, s);
            s = wmma_bf16(qf1, b1, s);
            #pragma unroll
            for (int v = 0; v < 8; v++) {
                int r = mstrip * 16 + v + ((lane >> 4) << 3);
                scores[r * 224 + ct * 16 + (lane & 15)] = s[v];
            }
        }
    }
    __syncthreads();

    // ---- phase 2: masked softmax (one thread per row) ----
    if (tid < 32) {
        int r = tid, t = row0 + r;
        bf16_t* ar = attn + r * 224;
        if (t < T_) {
            int kl = klen[b], ql = qlen[b];
            float* sr = scores + r * 224;
            float mx = -3.4e38f;
            for (int c = 0; c < T_; c++) {
                float s = sr[c];
                if (c >= kl || c == t) s = NEGV;   // key mask + diagonal blinding
                sr[c] = s;
                mx = fmaxf(mx, s);
            }
            float sum = 0.0f;
            for (int c = 0; c < T_; c++) { float p = __expf(sr[c] - mx); sr[c] = p; sum += p; }
            float inv = (t < ql) ? (1.0f / sum) : 0.0f;   // q_mask post-softmax
            for (int c = 0; c < T_; c++) ar[c] = (bf16_t)(sr[c] * inv);
            for (int c = T_; c < 224; c++) ar[c] = (bf16_t)0.0f;
        } else {
            for (int c = 0; c < 224; c++) ar[c] = (bf16_t)0.0f;
        }
    }
    __syncthreads();

    // ---- phase 3: out = attn @ V + residual q_in (V frags direct from HBM) ----
    {
        const int mstrip = wave & 1;
        const int n0 = (wave >> 1) * 32;
        v8f acc[2] = {};
        for (int k0 = 0; k0 < 224; k0 += 32) {
            v16bf a = load_frag_a(attn, 224, mstrip * 16, k0, lane);
            #pragma unroll
            for (int sn = 0; sn < 2; sn++) {
                v16bf bv = load_frag_bt(Vp, TPAD, n0 + sn * 16, k0, lane);
                acc[sn] = wmma_bf16(a, bv, acc[sn]);
            }
        }
        #pragma unroll
        for (int sn = 0; sn < 2; sn++)
            #pragma unroll
            for (int v = 0; v < 8; v++) {
                int rl = mstrip * 16 + v + ((lane >> 4) << 3);
                int t = row0 + rl;
                if (t < T_) {
                    int d = n0 + sn * 16 + (lane & 15);
                    int e = h * DH_ + d;
                    size_t gi = (size_t)(b * T_ + t) * E_ + e;
                    float val = acc[sn][v] + queries[gi] + pos_q[t * E_ + e] * SQRT_E;
                    out1[gi]  = val;
                    out1b[gi] = (bf16_t)val;   // bf16 copy feeds FFN1's A path
                }
            }
    }
}

// ---------------------------------------------------------------------------
// Kernel 5: FFN1  h1 = relu(out1 @ fw1)   (M=51200, N=2048, K=512)
// ---------------------------------------------------------------------------
__global__ __launch_bounds__(256) void ffn1_gemm(
        const bf16_t* __restrict__ out1b, const bf16_t* __restrict__ f1t,
        bf16_t* __restrict__ h1) {
    __shared__ __align__(16) bf16_t As[2][128 * 32];
    __shared__ __align__(16) bf16_t Bs[2][64 * 32];
    const int rowBase = blockIdx.y * 128;
    const int nBase = blockIdx.x * 64;

    GEMM_PIPELINE(out1b, E_, f1t, E_, E_)

    #pragma unroll
    for (int sm = 0; sm < 2; sm++)
        #pragma unroll
        for (int sn = 0; sn < 2; sn++)
            #pragma unroll
            for (int v = 0; v < 8; v++) {
                int r = rowBase + wr * 32 + sm * 16 + v + ((lane >> 4) << 3);
                int c = nBase + wc * 32 + sn * 16 + (lane & 15);
                float x = acc[sm][sn][v];
                x = x > 0.0f ? x : 0.0f;
                h1[(size_t)r * FF_ + c] = (bf16_t)x;
            }
}

// ---------------------------------------------------------------------------
// Kernel 6: FFN2  final = out1 + h1 @ fw2   (M=51200, N=512, K=2048)
// ---------------------------------------------------------------------------
__global__ __launch_bounds__(256) void ffn2_gemm(
        const bf16_t* __restrict__ h1, const bf16_t* __restrict__ f2t,
        const float* __restrict__ out1, float* __restrict__ fin) {
    __shared__ __align__(16) bf16_t As[2][128 * 32];
    __shared__ __align__(16) bf16_t Bs[2][64 * 32];
    const int rowBase = blockIdx.y * 128;
    const int nBase = blockIdx.x * 64;

    GEMM_PIPELINE(h1, FF_, f2t, FF_, FF_)

    #pragma unroll
    for (int sm = 0; sm < 2; sm++)
        #pragma unroll
        for (int sn = 0; sn < 2; sn++)
            #pragma unroll
            for (int v = 0; v < 8; v++) {
                int r = rowBase + wr * 32 + sm * 16 + v + ((lane >> 4) << 3);
                int c = nBase + wc * 32 + sn * 16 + (lane & 15);
                size_t gi = (size_t)r * E_ + c;
                fin[gi] = acc[sm][sn][v] + out1[gi];
            }
}

// ---------------------------------------------------------------------------
// Kernel 7: mean over T -> d_out (B,1,E)
// ---------------------------------------------------------------------------
__global__ void mean_reduce(const float* __restrict__ fin, float* __restrict__ dout) {
    int i = blockIdx.x * blockDim.x + threadIdx.x;
    if (i >= B_ * E_) return;
    int b = i / E_, e = i % E_;
    const float* p = fin + (size_t)b * T_ * E_ + e;
    float s = 0.0f;
    for (int t = 0; t < T_; t++) s += p[t * E_];
    dout[i] = s * (1.0f / T_);
}

// ---------------------------------------------------------------------------
extern "C" void kernel_launch(void* const* d_in, const int* in_sizes, int n_in,
                              void* d_out, int out_size, void* d_ws, size_t ws_size,
                              hipStream_t stream) {
    (void)in_sizes; (void)n_in; (void)out_size; (void)ws_size;
    const float* queries = (const float*)d_in[0];
    const float* keys    = (const float*)d_in[1];
    const int*   qlen    = (const int*)d_in[2];
    const int*   klen    = (const int*)d_in[3];
    const float* pos_q   = (const float*)d_in[4];
    const float* pos_k   = (const float*)d_in[5];
    const float* W_Q     = (const float*)d_in[6];
    const float* W_K     = (const float*)d_in[7];
    const float* W_V     = (const float*)d_in[8];
    const float* fw1     = (const float*)d_in[9];
    const float* fw2     = (const float*)d_in[10];
    float* dout = (float*)d_out;

    char* ws = (char*)d_ws;
    size_t off = 0;
    auto take = [&](size_t bytes) -> void* {
        void* p = ws + off;
        off += (bytes + 255) & ~(size_t)255;
        return p;
    };
    const size_t EW = (size_t)E_ * E_;
    const size_t FW = (size_t)E_ * FF_;
    bf16_t* wqt   = (bf16_t*)take(EW * 2);
    bf16_t* wkt   = (bf16_t*)take(EW * 2);
    bf16_t* wvt   = (bf16_t*)take(EW * 2);
    bf16_t* f1t   = (bf16_t*)take(FW * 2);
    bf16_t* f2t   = (bf16_t*)take(FW * 2);
    bf16_t* qinb  = (bf16_t*)take((size_t)M_ * E_ * 2);
    bf16_t* kinb  = (bf16_t*)take((size_t)M_ * E_ * 2);
    bf16_t* Qh    = (bf16_t*)take((size_t)B_ * H_ * TPAD * DH_ * 2);
    bf16_t* Kh    = (bf16_t*)take((size_t)B_ * H_ * TPAD * DH_ * 2);
    bf16_t* VtG   = (bf16_t*)take((size_t)B_ * H_ * DH_ * TPAD * 2);
    float*  out1  = (float*)take((size_t)M_ * E_ * 4);
    bf16_t* out1b = (bf16_t*)take((size_t)M_ * E_ * 2);
    bf16_t* h1    = (bf16_t*)take((size_t)M_ * FF_ * 2);
    float*  fin   = (float*)take((size_t)M_ * E_ * 4);

    prep_weights<<<1024, 256, 0, stream>>>(W_Q, W_K, W_V, fw1, fw2, wqt, wkt, wvt, f1t, f2t);
    prep_inputs<<<4096, 256, 0, stream>>>(queries, keys, pos_q, pos_k, qinb, kinb);
    proj_gemm<<<dim3(8, 400, 3), 256, 0, stream>>>(qinb, kinb, wqt, wkt, wvt, Qh, Kh, VtG);
    attn_kernel<<<dim3(7, H_, B_), 128, 0, stream>>>(Qh, Kh, VtG, qlen, klen,
                                                     queries, pos_q, out1, out1b);
    ffn1_gemm<<<dim3(FF_ / 64, 400), 256, 0, stream>>>(out1b, f1t, h1);
    ffn2_gemm<<<dim3(E_ / 64, 400), 256, 0, stream>>>(h1, f2t, out1, fin);
    mean_reduce<<<(B_ * E_ + 255) / 256, 256, 0, stream>>>(fin, dout);
}
